// GroupingLayer_37220186587445
// MI455X (gfx1250) — matrix operand order repaired and checked by reading, not compiled, not run on hardware
//
#include <hip/hip_runtime.h>
#include <cstdint>
#include <cstddef>

// ---------------- problem constants ----------------
#define BB    16
#define HH    56
#define WW    56
#define DD    384
#define HIDD  1152
#define NIN   3136      // 56*56
#define NOUT  784       // 28*28
#define HO    28
#define WO    28
#define MIN_  12544     // BB*NOUT
#define MKV   50176     // BB*NIN

typedef __attribute__((ext_vector_type(16))) __bf16          v16bf;
typedef __attribute__((ext_vector_type(16))) unsigned short  v16us;
typedef __attribute__((ext_vector_type(8)))  float           v8f;
typedef unsigned short ushort_t;

struct U16x16 { uint4 a, b; };   // 32B staging for a 16-element bf16 fragment

static __device__ __forceinline__ unsigned short f2bf(float f) {
    unsigned u = __builtin_bit_cast(unsigned, f);
    u += 0x7FFFu + ((u >> 16) & 1u);          // round-to-nearest-even
    return (unsigned short)(u >> 16);
}
static __device__ __forceinline__ float b2f(unsigned short h) {
    return __builtin_bit_cast(float, (unsigned)h << 16);
}
static __device__ __forceinline__ unsigned pack2(unsigned short lo, unsigned short hi) {
    return (unsigned)lo | ((unsigned)hi << 16);
}
static __device__ __forceinline__ float gelu_f(float x) {
    return 0.5f * x * (1.0f + erff(x * 0.70710678118654752f));
}

// ---------------------------------------------------------------------------
// Tiled WMMA GEMM:  C[M,N] = A[M,K] @ B[K,N] + bias[N]
//   A  : bf16, row-major (M,K)
//   BT : bf16, TRANSPOSED weights (N,K) row-major -> contiguous-K loads
// AMODE: 0 = plain A;  1 = conv2x2-s2 gather from bf16 X_in (M=12544, K=1536)
// EPI:   0 = fp32 out;  1 = bf16 out;  2 = exact-GELU + bf16 out
// Block: 128 threads (4 waves), tile 64x64, K-step 32. Pure copy in/out of LDS;
// all fragments are contiguous 2x b128 LDS loads.
// ---------------------------------------------------------------------------
template <int AMODE, int EPI>
__global__ __launch_bounds__(128)
void gemm_wmma(const ushort_t* __restrict__ A, const ushort_t* __restrict__ BT,
               const float* __restrict__ bias, void* __restrict__ Cout,
               int M, int N, int Kd)
{
    __shared__ __align__(16) unsigned short As[64][40];   // [row][k], 80B stride
    __shared__ __align__(16) unsigned short Bs[64][40];   // [col][k], 80B stride

    const int tid  = threadIdx.x;
    const int lane = tid & 31;
    const int wave = tid >> 5;
    const int wm   = wave >> 1;
    const int wn   = wave & 1;
    const int bm   = blockIdx.y * 64;
    const int bn   = blockIdx.x * 64;

    v8f zero = {0.f,0.f,0.f,0.f,0.f,0.f,0.f,0.f};
    v8f acc[2][2];
    acc[0][0] = zero; acc[0][1] = zero; acc[1][0] = zero; acc[1][1] = zero;

    // B-loader coords (fixed per thread)
    const int bcol = tid & 63;
    const int bkb  = (tid >> 6) * 16;
    const ushort_t* btp = BT + (size_t)(bn + bcol) * Kd + bkb;

    for (int k0 = 0; k0 < Kd; k0 += 32) {
        // ---- A tile: 64 rows x 32 k (bf16), 8 elems per slot ----
        #pragma unroll
        for (int i = 0; i < 2; ++i) {
            int s   = tid + i * 128;        // 0..255
            int row = s >> 2;
            int c8  = (s & 3) * 8;
            const ushort_t* p;
            if (AMODE == 0) {
                p = A + (size_t)(bm + row) * Kd + k0 + c8;
            } else {
                int n  = bm + row;
                int b  = n / NOUT, q = n % NOUT;
                int io = q / WO,   jo = q % WO;
                int tap = k0 / DD;                 // constant across the slice
                int cc  = (k0 % DD) + c8;
                int dy = tap >> 1, dx = tap & 1;
                int rin = (2 * io + dy) * WW + (2 * jo + dx);
                p = A + ((size_t)b * NIN + rin) * DD + cc;
            }
            *reinterpret_cast<uint4*>(&As[row][c8]) = *reinterpret_cast<const uint4*>(p);
        }
        // ---- B tile (transposed layout): 64 cols x 32 k ----
        {
            const ushort_t* p = btp + k0;
            uint4 lo = *reinterpret_cast<const uint4*>(p);
            uint4 hi = *reinterpret_cast<const uint4*>(p + 8);
            *reinterpret_cast<uint4*>(&Bs[bcol][bkb])     = lo;
            *reinterpret_cast<uint4*>(&Bs[bcol][bkb + 8]) = hi;
        }
        __syncthreads();

        // ---- fragments: each is two contiguous b128 LDS loads ----
        const int mr = lane & 15;
        const int kA = (lane < 16) ? 0 : 8;     // A: K 0-7/16-23 vs 8-15/24-31
        const int kB = (lane < 16) ? 0 : 8;     // B: K 0-7/16-23 vs 8-15/24-31
        v16us afr[2], bfr[2];
        #pragma unroll
        for (int mt = 0; mt < 2; ++mt) {
            const unsigned short* rp = &As[wm * 32 + mt * 16 + mr][0];
            U16x16 t;
            t.a = *reinterpret_cast<const uint4*>(rp + kA);
            t.b = *reinterpret_cast<const uint4*>(rp + kA + 16);
            afr[mt] = __builtin_bit_cast(v16us, t);
        }
        #pragma unroll
        for (int nt = 0; nt < 2; ++nt) {
            const unsigned short* cp = &Bs[wn * 32 + nt * 16 + mr][0];
            U16x16 t;
            t.a = *reinterpret_cast<const uint4*>(cp + kB);
            t.b = *reinterpret_cast<const uint4*>(cp + kB + 16);
            bfr[nt] = __builtin_bit_cast(v16us, t);
        }
        #pragma unroll
        for (int mt = 0; mt < 2; ++mt)
            #pragma unroll
            for (int nt = 0; nt < 2; ++nt)
                acc[mt][nt] = __builtin_amdgcn_wmma_f32_16x16x32_bf16(
                    false, __builtin_bit_cast(v16bf, afr[mt]),
                    false, __builtin_bit_cast(v16bf, bfr[nt]),
                    (short)0, acc[mt][nt], false, false);
        __syncthreads();
    }

    // ---- epilogue (C/D layout: col = lane&15, rows r / r+8) ----
    const int mr    = lane & 15;
    const int rbase = (lane < 16) ? 0 : 8;
    #pragma unroll
    for (int mt = 0; mt < 2; ++mt) {
        #pragma unroll
        for (int nt = 0; nt < 2; ++nt) {
            int   col = bn + wn * 32 + nt * 16 + mr;
            float bv  = bias[col];
            #pragma unroll
            for (int r = 0; r < 8; ++r) {
                int   row = bm + wm * 32 + mt * 16 + rbase + r;
                float v   = acc[mt][nt][r] + bv;
                if (EPI == 2) v = gelu_f(v);
                if (EPI == 0)
                    ((float*)Cout)[(size_t)row * N + col] = v;
                else
                    ((ushort_t*)Cout)[(size_t)row * N + col] = f2bf(v);
            }
        }
    }
}

// ---------------------------------------------------------------------------
// fp32 -> bf16 bulk convert (8 elements / thread)
// ---------------------------------------------------------------------------
__global__ __launch_bounds__(256)
void cvt_bf16_kernel(const float* __restrict__ in, ushort_t* __restrict__ out, size_t n8)
{
    size_t i = (size_t)blockIdx.x * 256 + threadIdx.x;
    if (i >= n8) return;
    const float4* p = reinterpret_cast<const float4*>(in) + i * 2;
    float4 a = p[0], b = p[1];
    uint4 o;
    o.x = pack2(f2bf(a.x), f2bf(a.y));
    o.y = pack2(f2bf(a.z), f2bf(a.w));
    o.z = pack2(f2bf(b.x), f2bf(b.y));
    o.w = pack2(f2bf(b.z), f2bf(b.w));
    reinterpret_cast<uint4*>(out)[i] = o;
}

// W (K,N) fp32 -> WT (N,K) bf16
__global__ __launch_bounds__(256)
void wtrans_kernel(const float* __restrict__ W, ushort_t* __restrict__ WT, int K, int N)
{
    int idx = blockIdx.x * 256 + threadIdx.x;
    if (idx >= K * N) return;
    int k = idx % K, n = idx / K;
    WT[idx] = f2bf(W[(size_t)k * N + n]);
}

// conv_w (OIHW D,D,2,2) -> WcT (384, 1536) bf16 : WcT[o][tap*384+c] = w[o,c,dy,dx]
__global__ __launch_bounds__(256)
void convwt_kernel(const float* __restrict__ cw, ushort_t* __restrict__ WcT)
{
    int idx = blockIdx.x * 256 + threadIdx.x;       // = o*1536 + kk
    if (idx >= DD * 1536) return;
    int kk  = idx % 1536;
    int o   = idx / 1536;
    int c   = kk % DD;
    int tap = kk / DD;
    int dy = tap >> 1, dx = tap & 1;
    WcT[idx] = f2bf(cw[(((size_t)o * DD + c) * 2 + dy) * 2 + dx]);
}

__global__ __launch_bounds__(32)
void biasmean_kernel(const float* __restrict__ rb, float* __restrict__ out)
{
    int lane = threadIdx.x;
    float v = (lane < 25) ? rb[lane] : 0.f;
    #pragma unroll
    for (int off = 16; off > 0; off >>= 1) v += __shfl_xor(v, off, 32);
    if (lane == 0) out[0] = v / 25.f;
}

// ---------------------------------------------------------------------------
// out = (base?base:0) + LN(in)*sc + bi ; optional bf16 mirror obf   (D=384)
// ---------------------------------------------------------------------------
__global__ __launch_bounds__(128)
void add_ln_kernel(const float* __restrict__ in, const float* __restrict__ base,
                   float* __restrict__ out, ushort_t* __restrict__ obf,
                   const float* __restrict__ sc, const float* __restrict__ bi)
{
    __shared__ float red[128];
    const int row = blockIdx.x, tid = threadIdx.x;
    const float* ip = in + (size_t)row * DD;
    float x0 = ip[tid], x1 = ip[tid + 128], x2 = ip[tid + 256];

    red[tid] = x0 + x1 + x2;
    __syncthreads();
    for (int o = 64; o > 0; o >>= 1) { if (tid < o) red[tid] += red[tid + o]; __syncthreads(); }
    float mean = red[0] * (1.f / DD);
    __syncthreads();

    float d0 = x0 - mean, d1 = x1 - mean, d2 = x2 - mean;
    red[tid] = d0 * d0 + d1 * d1 + d2 * d2;
    __syncthreads();
    for (int o = 64; o > 0; o >>= 1) { if (tid < o) red[tid] += red[tid + o]; __syncthreads(); }
    float r = rsqrtf(red[0] * (1.f / DD) + 1e-5f);

    const float* bp = base ? base + (size_t)row * DD : nullptr;
    float y0 = (bp ? bp[tid]       : 0.f) + d0 * r * sc[tid]       + bi[tid];
    float y1 = (bp ? bp[tid + 128] : 0.f) + d1 * r * sc[tid + 128] + bi[tid + 128];
    float y2 = (bp ? bp[tid + 256] : 0.f) + d2 * r * sc[tid + 256] + bi[tid + 256];

    float* op = out + (size_t)row * DD;
    op[tid] = y0; op[tid + 128] = y1; op[tid + 256] = y2;
    if (obf) {
        ushort_t* bo = obf + (size_t)row * DD;
        bo[tid] = f2bf(y0); bo[tid + 128] = f2bf(y1); bo[tid + 256] = f2bf(y2);
    }
}

// ---------------------------------------------------------------------------
// Sparse local attention: one wave32 per (b,q). <=9 neighbor logits -> softmax
// -> new_centers (fp32); compact weights to Wcomp (B,784,9).
// ---------------------------------------------------------------------------
__global__ __launch_bounds__(256)
void attn_kernel(const float* __restrict__ Q, const ushort_t* __restrict__ Kb,
                 const ushort_t* __restrict__ Vb, const float* __restrict__ biasS,
                 float* __restrict__ NC, float* __restrict__ Wcomp)
{
    const int wave = threadIdx.x >> 5, lane = threadIdx.x & 31;
    const int gw = blockIdx.x * 8 + wave;            // < 12544
    const int b = gw / NOUT, q = gw % NOUT;
    const int io = q / WO, jo = q % WO;
    const int yo = io * 2, xo = jo * 2;
    const float bias = biasS[0];

    float qr[12];
    const float* qp = Q + (size_t)gw * DD;
    #pragma unroll
    for (int t = 0; t < 12; ++t) qr[t] = qp[lane + 32 * t];

    float lg[9];
    #pragma unroll
    for (int j = 0; j < 9; ++j) {
        int dy = j / 3 - 1, dx = j % 3 - 1;
        int yi = yo + dy, xi = xo + dx;              // upper bounds always ok
        if (yi >= 0 && xi >= 0) {
            const ushort_t* kp = Kb + ((size_t)b * NIN + yi * WW + xi) * DD;
            float p = 0.f;
            #pragma unroll
            for (int t = 0; t < 12; ++t) p += qr[t] * b2f(kp[lane + 32 * t]);
            #pragma unroll
            for (int off = 16; off > 0; off >>= 1) p += __shfl_xor(p, off, 32);
            lg[j] = p + bias;
        } else {
            lg[j] = -1e30f;
        }
    }
    float m = lg[0];
    #pragma unroll
    for (int j = 1; j < 9; ++j) m = fmaxf(m, lg[j]);
    float w[9], s = 0.f;
    #pragma unroll
    for (int j = 0; j < 9; ++j) { w[j] = __expf(lg[j] - m); s += w[j]; }
    float inv = 1.f / s;
    #pragma unroll
    for (int j = 0; j < 9; ++j) w[j] *= inv;

    float o[12];
    #pragma unroll
    for (int t = 0; t < 12; ++t) o[t] = 0.f;
    #pragma unroll
    for (int j = 0; j < 9; ++j) {
        int dy = j / 3 - 1, dx = j % 3 - 1;
        int yi = yo + dy, xi = xo + dx;
        if (yi >= 0 && xi >= 0) {
            const ushort_t* vp = Vb + ((size_t)b * NIN + yi * WW + xi) * DD;
            #pragma unroll
            for (int t = 0; t < 12; ++t) o[t] += w[j] * b2f(vp[lane + 32 * t]);
        }
    }
    float* op = NC + (size_t)gw * DD;
    #pragma unroll
    for (int t = 0; t < 12; ++t) op[lane + 32 * t] = o[t];

    if (lane == 0) {
        float* wp = Wcomp + (size_t)gw * 9;
        #pragma unroll
        for (int j = 0; j < 9; ++j) wp[j] = w[j];
    }
}

// ---------------------------------------------------------------------------
// colsum[b,k] = sum_q A_ups[b,q,k]   (<=4 contributors per input token)
// ---------------------------------------------------------------------------
__global__ __launch_bounds__(256)
void colsum_kernel(const float* __restrict__ Wcomp, float* __restrict__ colsum)
{
    int idx = blockIdx.x * 256 + threadIdx.x;
    if (idx >= BB * NIN) return;
    int b = idx / NIN, k = idx % NIN;
    int yi = k / WW, xi = k % WW;
    float s = 0.f;
    for (int yo = yi - 1; yo <= yi + 1; ++yo) {
        if (yo < 0 || yo > 54 || (yo & 1)) continue;
        for (int xo = xi - 1; xo <= xi + 1; ++xo) {
            if (xo < 0 || xo > 54 || (xo & 1)) continue;
            int q = (yo >> 1) * WO + (xo >> 1);
            s += Wcomp[((size_t)b * NOUT + q) * 9 + (yi - yo + 1) * 3 + (xi - xo + 1)];
        }
    }
    colsum[idx] = s;
}

// ---------------------------------------------------------------------------
// Materialize dense A_ups / A_down (float4 stores; 56 % 4 == 0)
// ---------------------------------------------------------------------------
__global__ __launch_bounds__(256)
void expand_kernel(const float* __restrict__ Wcomp, const float* __restrict__ colsum,
                   float* __restrict__ ups, float* __restrict__ down)
{
    size_t idx = (size_t)blockIdx.x * 256 + threadIdx.x;       // quad index
    const size_t TOTQ = (size_t)BB * NOUT * (NIN / 4);
    if (idx >= TOTQ) return;
    int kq  = (int)(idx % (NIN / 4));
    int rem = (int)(idx / (NIN / 4));
    int q = rem % NOUT, b = rem / NOUT;
    int k4 = kq * 4;
    int yi = k4 / WW, xib = k4 % WW;
    int io = q / WO, jo = q % WO;
    int yo = 2 * io, xo = 2 * jo;
    int dy = yi - yo;
    const float* wrow = Wcomp + ((size_t)b * NOUT + q) * 9;
    const float* crow = colsum + (size_t)b * NIN + k4;
    float ua[4], da[4];
    #pragma unroll
    for (int t = 0; t < 4; ++t) {
        int dx = xib + t - xo;
        float v = 0.f;
        if (dy >= -1 && dy <= 1 && dx >= -1 && dx <= 1)
            v = wrow[(dy + 1) * 3 + (dx + 1)];
        ua[t] = v;
        da[t] = v / (1e-10f + crow[t]);
    }
    size_t off = ((size_t)b * NOUT + q) * NIN + k4;
    *reinterpret_cast<float4*>(ups  + off) = make_float4(ua[0], ua[1], ua[2], ua[3]);
    *reinterpret_cast<float4*>(down + off) = make_float4(da[0], da[1], da[2], da[3]);
}

// ---------------------------------------------------------------------------
extern "C" void kernel_launch(void* const* d_in, const int* in_sizes, int n_in,
                              void* d_out, int out_size, void* d_ws, size_t ws_size,
                              hipStream_t stream)
{
    const float* X_in    = (const float*)d_in[0];
    const float* conv_w  = (const float*)d_in[1];
    const float* conv_b  = (const float*)d_in[2];
    const float* ln_i_s  = (const float*)d_in[3];
    const float* ln_i_b  = (const float*)d_in[4];
    const float* qw      = (const float*)d_in[5];
    const float* qb      = (const float*)d_in[6];
    const float* kw      = (const float*)d_in[7];
    const float* kb      = (const float*)d_in[8];
    const float* vw      = (const float*)d_in[9];
    const float* vb      = (const float*)d_in[10];
    const float* ln_o_s  = (const float*)d_in[11];
    const float* ln_o_b  = (const float*)d_in[12];
    const float* w1      = (const float*)d_in[13];
    const float* b1      = (const float*)d_in[14];
    const float* w2      = (const float*)d_in[15];
    const float* b2      = (const float*)d_in[16];
    const float* mn_s    = (const float*)d_in[17];
    const float* mn_b    = (const float*)d_in[18];
    const float* rel     = (const float*)d_in[19];
    (void)in_sizes; (void)n_in; (void)out_size; (void)ws_size;

    // ---- workspace carve-up ----
    char* ws = (char*)d_ws;
    size_t off = 0;
    auto take = [&](size_t bytes) { char* p = ws + off; off += (bytes + 255) & ~(size_t)255; return p; };
    ushort_t* XinBf  = (ushort_t*)take((size_t)MKV  * DD * 2);
    ushort_t* Kbuf   = (ushort_t*)take((size_t)MKV  * DD * 2);
    ushort_t* Vbuf   = (ushort_t*)take((size_t)MKV  * DD * 2);
    float*    Xout   = (float*)   take((size_t)MIN_ * DD * 4);
    ushort_t* XoutBf = (ushort_t*)take((size_t)MIN_ * DD * 2);
    float*    Qbuf   = (float*)   take((size_t)MIN_ * DD * 4);
    float*    Tmp    = (float*)   take((size_t)MIN_ * DD * 4);
    ushort_t* Hbuf   = (ushort_t*)take((size_t)MIN_ * HIDD * 2);
    ushort_t* qwT    = (ushort_t*)take((size_t)DD * DD * 2);
    ushort_t* kwT    = (ushort_t*)take((size_t)DD * DD * 2);
    ushort_t* vwT    = (ushort_t*)take((size_t)DD * DD * 2);
    ushort_t* w1T    = (ushort_t*)take((size_t)DD * HIDD * 2);
    ushort_t* w2T    = (ushort_t*)take((size_t)DD * HIDD * 2);
    ushort_t* WcT    = (ushort_t*)take((size_t)DD * 1536 * 2);
    float*    Wcomp  = (float*)   take((size_t)MIN_ * 9 * 4);
    float*    Csum   = (float*)   take((size_t)BB * NIN * 4);
    float*    BiasS  = (float*)   take(256);

    float* outX    = (float*)d_out;                              // (B,784,384)
    float* outDown = outX + (size_t)MIN_ * DD;                   // (B,784,3136)
    float* outUps  = outDown + (size_t)BB * NOUT * NIN;          // (B,784,3136)

    // ---- one-time precision/layout conversions ----
    cvt_bf16_kernel<<<(unsigned)(((size_t)MKV * DD / 8 + 255) / 256), 256, 0, stream>>>(
        X_in, XinBf, (size_t)MKV * DD / 8);
    wtrans_kernel<<<(DD * DD + 255) / 256, 256, 0, stream>>>(qw, qwT, DD, DD);
    wtrans_kernel<<<(DD * DD + 255) / 256, 256, 0, stream>>>(kw, kwT, DD, DD);
    wtrans_kernel<<<(DD * DD + 255) / 256, 256, 0, stream>>>(vw, vwT, DD, DD);
    wtrans_kernel<<<(DD * HIDD + 255) / 256, 256, 0, stream>>>(w1, w1T, DD, HIDD);   // -> (1152,384)
    wtrans_kernel<<<(DD * HIDD + 255) / 256, 256, 0, stream>>>(w2, w2T, HIDD, DD);   // -> (384,1152)
    convwt_kernel<<<(DD * 1536 + 255) / 256, 256, 0, stream>>>(conv_w, WcT);         // -> (384,1536)
    biasmean_kernel<<<1, 32, 0, stream>>>(rel, BiasS);

    // ---- K, V projections (bf16 outputs) ----
    gemm_wmma<0, 1><<<dim3(DD / 64, MKV / 64), 128, 0, stream>>>(XinBf, kwT, kb, Kbuf, MKV, DD, DD);
    gemm_wmma<0, 1><<<dim3(DD / 64, MKV / 64), 128, 0, stream>>>(XinBf, vwT, vb, Vbuf, MKV, DD, DD);

    // ---- init_proj conv (gathered GEMM, K=1536) then LN ----
    gemm_wmma<1, 0><<<dim3(DD / 64, MIN_ / 64), 128, 0, stream>>>(XinBf, WcT, conv_b, Tmp, MIN_, DD, 1536);
    add_ln_kernel<<<MIN_, 128, 0, stream>>>(Tmp, nullptr, Xout, XoutBf, ln_i_s, ln_i_b);

    // ---- 3 grouping iterations ----
    for (int it = 0; it < 3; ++it) {
        gemm_wmma<0, 0><<<dim3(DD / 64, MIN_ / 64), 128, 0, stream>>>(XoutBf, qwT, qb, Qbuf, MIN_, DD, DD);
        attn_kernel<<<MIN_ / 8, 256, 0, stream>>>(Qbuf, Kbuf, Vbuf, BiasS, Tmp, Wcomp);
        add_ln_kernel<<<MIN_, 128, 0, stream>>>(Tmp, Xout, Xout, XoutBf, ln_o_s, ln_o_b);

        gemm_wmma<0, 2><<<dim3(HIDD / 64, MIN_ / 64), 128, 0, stream>>>(XoutBf, w1T, b1, Hbuf, MIN_, HIDD, DD);
        gemm_wmma<0, 0><<<dim3(DD / 64, MIN_ / 64), 128, 0, stream>>>(Hbuf, w2T, b2, Tmp, MIN_, DD, HIDD);
        float*    dst  = (it == 2) ? outX : Xout;
        ushort_t* dstb = (it == 2) ? nullptr : XoutBf;
        add_ln_kernel<<<MIN_, 128, 0, stream>>>(Tmp, Xout, dst, dstb, mn_s, mn_b);
    }

    // ---- materialize A_ups / A_down from final iteration's weights ----
    colsum_kernel<<<(BB * NIN + 255) / 256, 256, 0, stream>>>(Wcomp, Csum);
    const size_t TOTQ = (size_t)BB * NOUT * (NIN / 4);
    expand_kernel<<<(unsigned)((TOTQ + 255) / 256), 256, 0, stream>>>(Wcomp, Csum, outUps, outDown);
}
